// MultiHeadAttention_87411174408722
// MI455X (gfx1250) — matrix operand order repaired and checked by reading, not compiled, not run on hardware
//
#include <hip/hip_runtime.h>
#include <hip/hip_bf16.h>
#include <stdint.h>

// ---------------------------------------------------------------------------
// MultiHeadAttention (with the source's attn@scores bug), MI455X / gfx1250.
// bf16 WMMA (v_wmma_f32_16x16x32_bf16) for all GEMMs, f32 softmax,
// async LDS staging (global_load_async_to_lds_b128) in the attention stage.
// ---------------------------------------------------------------------------

#define BB 64
#define SS 128
#define EE 2048
#define HH 16
#define HD 128
#define MM (BB * SS)      // 8192 rows in the projection GEMMs

typedef __bf16 v16bf __attribute__((ext_vector_type(16)));
typedef float  v8f   __attribute__((ext_vector_type(8)));

struct __align__(16) U128 { unsigned x, y, z, w; };
union FragU { U128 q[2]; v16bf v; };

__device__ __forceinline__ unsigned short f2bf(float f) {
  unsigned u = __float_as_uint(f);
  u += 0x7fffu + ((u >> 16) & 1u);          // round-to-nearest-even
  return (unsigned short)(u >> 16);
}

// A-fragment (16x32 bf16, M x K): lane<16 holds row M=lane, K = {k0..k0+7, k0+16..k0+23};
// lane>=16 holds K = {k0+8..k0+15, k0+24..k0+31}.  (ISA 7.12.2, 16-bit A 16x32)
__device__ __forceinline__ v16bf load_a_frag(const unsigned short* base, int row0, int k0, int ld) {
  const int lane = threadIdx.x & 31;
  const int r = lane & 15, hi = lane >> 4;
  const unsigned short* p = base + (size_t)(row0 + r) * ld + k0 + hi * 8;
  FragU f;
  f.q[0] = *(const U128*)(p);
  f.q[1] = *(const U128*)(p + 16);
  return f.v;
}

// B-fragment (32x16 bf16, K x N), source stored as rows of W (N x K, K contiguous):
// lane<16 holds col N=lane, K = {k0..k0+7, k0+16..k0+23}; lane>=16 holds K = {k0+8..k0+15, k0+24..k0+31}.
// (half of the ISA's 64x16 16-bit B layout)
__device__ __forceinline__ v16bf load_b_frag(const unsigned short* base, int col0, int k0, int ld) {
  const int lane = threadIdx.x & 31;
  const int c = lane & 15, hi = lane >> 4;
  const unsigned short* p = base + (size_t)(col0 + c) * ld + k0 + hi * 8;
  FragU f;
  f.q[0] = *(const U128*)(p);
  f.q[1] = *(const U128*)(p + 16);
  return f.v;
}

#define WMMA_BF16(a, b, c) \
  __builtin_amdgcn_wmma_f32_16x16x32_bf16(false, (a), false, (b), (short)0, (c), false, false)

// ---------------------------------------------------------------------------
// fp32 -> bf16 conversion
// ---------------------------------------------------------------------------
__global__ __launch_bounds__(256) void cvt_bf16_kernel(const float* __restrict__ src,
                                                       unsigned short* __restrict__ dst, int n) {
  int i = blockIdx.x * 256 + threadIdx.x;
  if (i < n) dst[i] = f2bf(src[i]);
}

// ---------------------------------------------------------------------------
// C = A(bf16 MxK row-major) @ W(bf16 NxK row-major)^T + bias.
// Workgroup: 128x128 tile, 8 waves of 32 lanes; wave = 32x64 (2x4 wmma tiles).
// All operands are L2-resident (56 MB bf16 << 192 MB L2) -> fragments loaded
// straight from global memory (global_load_b128), no LDS staging needed.
// mode 0: f32 row-major [M][N]   (final projection -> d_out)
// mode 1: bf16 scattered into [B,H,S,HD] (Q/K for attention)
// ---------------------------------------------------------------------------
__global__ __launch_bounds__(256) void gemm_bf16_kernel(const unsigned short* __restrict__ A,
                                                        const unsigned short* __restrict__ W,
                                                        const float* __restrict__ bias,
                                                        void* __restrict__ out, int mode) {
  const int lane = threadIdx.x & 31;
  const int w    = threadIdx.x >> 5;
  const int m0   = blockIdx.y * 128 + (w & 3) * 32;
  const int n0   = blockIdx.x * 128 + (w >> 2) * 64;

  v8f acc[2][4];
#pragma unroll
  for (int i = 0; i < 2; ++i)
#pragma unroll
    for (int j = 0; j < 4; ++j) acc[i][j] = {};

  for (int kk = 0; kk < EE; kk += 32) {
    v16bf a0 = load_a_frag(A, m0, kk, EE);
    v16bf a1 = load_a_frag(A, m0 + 16, kk, EE);
    v16bf b0 = load_b_frag(W, n0, kk, EE);
    v16bf b1 = load_b_frag(W, n0 + 16, kk, EE);
    v16bf b2 = load_b_frag(W, n0 + 32, kk, EE);
    v16bf b3 = load_b_frag(W, n0 + 48, kk, EE);
    acc[0][0] = WMMA_BF16(a0, b0, acc[0][0]);
    acc[0][1] = WMMA_BF16(a0, b1, acc[0][1]);
    acc[0][2] = WMMA_BF16(a0, b2, acc[0][2]);
    acc[0][3] = WMMA_BF16(a0, b3, acc[0][3]);
    acc[1][0] = WMMA_BF16(a1, b0, acc[1][0]);
    acc[1][1] = WMMA_BF16(a1, b1, acc[1][1]);
    acc[1][2] = WMMA_BF16(a1, b2, acc[1][2]);
    acc[1][3] = WMMA_BF16(a1, b3, acc[1][3]);
  }

  const int r8 = (lane >> 4) * 8;   // C layout: VGPR r, lane<16 -> M=r, lane>=16 -> M=8+r
  const int cn = lane & 15;
#pragma unroll
  for (int i = 0; i < 2; ++i)
#pragma unroll
    for (int j = 0; j < 4; ++j)
#pragma unroll
      for (int r = 0; r < 8; ++r) {
        const int gm = m0 + 16 * i + r8 + r;
        const int gn = n0 + 16 * j + cn;
        const float v = acc[i][j][r] + bias[gn];
        if (mode == 0) {
          ((float*)out)[(size_t)gm * EE + gn] = v;
        } else {
          const int b = gm >> 7, s = gm & 127, h = gn >> 7, d = gn & 127;
          ((unsigned short*)out)[(((size_t)b * HH + h) * SS + s) * HD + d] = f2bf(v);
        }
      }
}

// ---------------------------------------------------------------------------
// Attention core: one workgroup per (b,h).
// LDS plan (dynamic, base offset 0):
//   [0      , 32K) : Qs bf16 [128][128]   -> reused as Pb  (softmax(scores), bf16)
//   [32K    , 64K) : Ks bf16 [128][128]   -> reused as ScT (scores^T, bf16)
//   [64K    , ..)  : Scf f32 [128][132]   (scaled scores, softmax in f32)
// ---------------------------------------------------------------------------
#define SCP 132
#define ATTN_SMEM (65536 + SS * SCP * 4)

__global__ __launch_bounds__(256) void attn_kernel(const unsigned short* __restrict__ Qws,
                                                   const unsigned short* __restrict__ Kws,
                                                   unsigned short* __restrict__ out2) {
  extern __shared__ char smem[];
  unsigned short* Qs  = (unsigned short*)smem;
  unsigned short* Ks  = (unsigned short*)(smem + 32768);
  unsigned short* Pb  = (unsigned short*)smem;            // reuse after GEMM1
  unsigned short* ScT = (unsigned short*)(smem + 32768);  // reuse after GEMM1
  float* Scf = (float*)(smem + 65536);

  const int t  = threadIdx.x;
  const int bh = blockIdx.x;
  const int b  = bh >> 4;
  const int h  = bh & 15;

  // --- async stage Q,K (32 KB each) into LDS via the CDNA5 async path ---
  const char* Qg = (const char*)(Qws + (size_t)bh * SS * HD);
  const char* Kg = (const char*)(Kws + (size_t)bh * SS * HD);
#pragma unroll
  for (int i = 0; i < 8; ++i) {
    unsigned o = (unsigned)(t + i * 256) * 16u;
    asm volatile("global_load_async_to_lds_b128 %0, %1, off" ::"v"(o), "v"(Qg + o) : "memory");
    asm volatile("global_load_async_to_lds_b128 %0, %1, off" ::"v"(32768u + o), "v"(Kg + o) : "memory");
  }
  asm volatile("s_wait_asynccnt 0" ::: "memory");
  __syncthreads();

  const int lane = t & 31;
  const int w    = t >> 5;
  const int wm   = (w & 3) * 32;
  const int wn   = (w >> 2) * 64;
  const int r8   = (lane >> 4) * 8;
  const int cn   = lane & 15;

  // --- GEMM1: scores = Q @ K^T * (1/sqrt(HD)) ---
  {
    v8f acc[2][4];
#pragma unroll
    for (int i = 0; i < 2; ++i)
#pragma unroll
      for (int j = 0; j < 4; ++j) acc[i][j] = {};
#pragma unroll
    for (int kk = 0; kk < HD; kk += 32) {
      v16bf a0 = load_a_frag(Qs, wm, kk, HD);
      v16bf a1 = load_a_frag(Qs, wm + 16, kk, HD);
#pragma unroll
      for (int j = 0; j < 4; ++j) {
        v16bf bf = load_b_frag(Ks, wn + 16 * j, kk, HD);
        acc[0][j] = WMMA_BF16(a0, bf, acc[0][j]);
        acc[1][j] = WMMA_BF16(a1, bf, acc[1][j]);
      }
    }
    const float scale = 0.088388347648318447f;  // 1/sqrt(128)
#pragma unroll
    for (int i = 0; i < 2; ++i)
#pragma unroll
      for (int j = 0; j < 4; ++j)
#pragma unroll
        for (int r = 0; r < 8; ++r)
          Scf[(wm + 16 * i + r8 + r) * SCP + wn + 16 * j + cn] = acc[i][j][r] * scale;
  }
  __syncthreads();

  // --- row softmax in f32; emit P (bf16) and Sc^T (bf16, K-contiguous B operand) ---
  if (t < SS) {
    const int q = t;
    float mx = -3.402823466e38f;
    for (int k = 0; k < SS; ++k) mx = fmaxf(mx, Scf[q * SCP + k]);
    float sum = 0.f;
    for (int k = 0; k < SS; ++k) sum += __expf(Scf[q * SCP + k] - mx);
    const float inv = 1.0f / sum;
    for (int k = 0; k < SS; ++k) {
      const float sc = Scf[q * SCP + k];
      Pb[q * SS + k]  = f2bf(__expf(sc - mx) * inv);
      ScT[k * SS + q] = f2bf(sc);
    }
  }
  __syncthreads();

  // --- GEMM2 (source bug, faithful): out = softmax(scores) @ scores ---
  {
    v8f acc[2][4];
#pragma unroll
    for (int i = 0; i < 2; ++i)
#pragma unroll
      for (int j = 0; j < 4; ++j) acc[i][j] = {};
#pragma unroll
    for (int kk = 0; kk < SS; kk += 32) {
      v16bf a0 = load_a_frag(Pb, wm, kk, SS);
      v16bf a1 = load_a_frag(Pb, wm + 16, kk, SS);
#pragma unroll
      for (int j = 0; j < 4; ++j) {
        v16bf bf = load_b_frag(ScT, wn + 16 * j, kk, SS);
        acc[0][j] = WMMA_BF16(a0, bf, acc[0][j]);
        acc[1][j] = WMMA_BF16(a1, bf, acc[1][j]);
      }
    }
    // write out2[b][q][h*S + l] (bf16) — the transpose(0,2,1,3).reshape fused in
#pragma unroll
    for (int i = 0; i < 2; ++i)
#pragma unroll
      for (int j = 0; j < 4; ++j)
#pragma unroll
        for (int r = 0; r < 8; ++r) {
          const int q = wm + 16 * i + r8 + r;
          const int l = wn + 16 * j + cn;
          out2[(size_t)b * SS * EE + (size_t)q * EE + h * SS + l] = f2bf(acc[i][j][r]);
        }
  }
}

// ---------------------------------------------------------------------------
// launch
// ---------------------------------------------------------------------------
extern "C" void kernel_launch(void* const* d_in, const int* in_sizes, int n_in,
                              void* d_out, int out_size, void* d_ws, size_t ws_size,
                              hipStream_t stream) {
  (void)in_sizes; (void)n_in; (void)out_size; (void)ws_size;

  const float* x  = (const float*)d_in[0];
  const float* Wq = (const float*)d_in[1];
  const float* bq = (const float*)d_in[2];
  const float* Wk = (const float*)d_in[3];
  const float* bk = (const float*)d_in[4];
  // d_in[5] = Wv, d_in[6] = bv: the reference never uses V -> skipped.
  const float* Wo = (const float*)d_in[7];
  const float* bo = (const float*)d_in[8];

  char* ws = (char*)d_ws;
  unsigned short* xb  = (unsigned short*)(ws);              // 32 MB: x bf16, reused as out2 bf16
  unsigned short* w1b = (unsigned short*)(ws + 33554432);   //  8 MB: Wq bf16, reused for Wo bf16
  unsigned short* w2b = (unsigned short*)(ws + 41943040);   //  8 MB: Wk bf16
  unsigned short* qws = (unsigned short*)(ws + 50331648);   // 32 MB: Q bf16 [B,H,S,HD]
  unsigned short* kws = (unsigned short*)(ws + 83886080);   // 32 MB: K bf16 [B,H,S,HD]
  unsigned short* o2b = xb;                                 // alias (x dead after projections)

  hipFuncSetAttribute((const void*)attn_kernel,
                      hipFuncAttributeMaxDynamicSharedMemorySize, ATTN_SMEM);

  const int nx = MM * EE, nw = EE * EE;
  cvt_bf16_kernel<<<(nx + 255) / 256, 256, 0, stream>>>(x, xb, nx);
  cvt_bf16_kernel<<<(nw + 255) / 256, 256, 0, stream>>>(Wq, w1b, nw);
  cvt_bf16_kernel<<<(nw + 255) / 256, 256, 0, stream>>>(Wk, w2b, nw);

  dim3 ggrid(EE / 128, MM / 128);  // (16, 64)
  gemm_bf16_kernel<<<ggrid, 256, 0, stream>>>(xb, w1b, bq, qws, 1);
  gemm_bf16_kernel<<<ggrid, 256, 0, stream>>>(xb, w2b, bk, kws, 1);

  cvt_bf16_kernel<<<(nw + 255) / 256, 256, 0, stream>>>(Wo, w1b, nw);  // Wq slot now dead

  attn_kernel<<<BB * HH, 256, ATTN_SMEM, stream>>>(qws, kws, o2b);

  gemm_bf16_kernel<<<ggrid, 256, 0, stream>>>(o2b, w1b, bo, d_out, 0);
}
